// FullNet_6725918785902
// MI455X (gfx1250) — matrix operand order repaired and checked by reading, not compiled, not run on hardware
//
#include <hip/hip_runtime.h>
#include <hip/hip_bf16.h>
#include <math.h>

#define B_   2
#define C_   64
#define H_   112
#define W_   112
#define N_   (H_*W_)   /* 12544 */
#define S_   50
#define G_   120

typedef __attribute__((ext_vector_type(16))) __bf16 bf16x16;
typedef __attribute__((ext_vector_type(8)))  __bf16 bf16x8;
typedef __attribute__((ext_vector_type(8)))  float  f32x8;

// ---------------------------------------------------------------------------
// Prep: build WMMA-native bf16 layouts fxT/fyT = [b*N + pixel][64], and
// rank_scale[m] = mask2[m] ? 1/(||fy_m||+1e-12) : 0  (cosine ranking scale).
// ---------------------------------------------------------------------------
__global__ __launch_bounds__(256)
void prep_kernel(const float* __restrict__ img1, const float* __restrict__ img2,
                 const unsigned char* __restrict__ mask2,
                 __bf16* __restrict__ fxT, __bf16* __restrict__ fyT,
                 float* __restrict__ rscale)
{
    int t = blockIdx.x * blockDim.x + threadIdx.x;
    if (t >= B_ * N_) return;
    int b = t / N_, n = t - b * N_;
    const float* f1 = img1 + (size_t)b * C_ * N_ + n;
    const float* f2 = img2 + (size_t)b * C_ * N_ + n;
    __bf16* ax = fxT + (size_t)t * C_;
    __bf16* ay = fyT + (size_t)t * C_;
    float ss = 0.0f;
#pragma unroll 8
    for (int c = 0; c < C_; ++c) {
        float v1 = f1[(size_t)c * N_];
        float v2 = f2[(size_t)c * N_];
        ax[c] = (__bf16)v1;
        ay[c] = (__bf16)v2;
        ss = fmaf(v2, v2, ss);
    }
    rscale[t] = mask2[t] ? (1.0f / (sqrtf(ss) + 1e-12f)) : 0.0f;
}

// A fragment, 16x32 bf16 (ISA 7.12.2): lanes 0-15 hold K {k0+0..7, k0+16..23},
// lanes 16-31 hold K {k0+8..15, k0+24..31}, row = n0 + (lane&15).
__device__ __forceinline__ bf16x16 load_a_frag(const __bf16* base, int k0, int half)
{
    const int o1 = k0 + 8 * half;
    const int o2 = o1 + 16;
    bf16x8 lo = *(const bf16x8*)(base + o1);
    bf16x8 hi = *(const bf16x8*)(base + o2);
    bf16x16 r;
#pragma unroll
    for (int i = 0; i < 8; ++i) { r[i] = lo[i]; r[i + 8] = hi[i]; }
    return r;
}

// B fragment, 32x16 bf16: lanes 0-15 hold K {k0+0..7, k0+16..23}, lanes 16-31 hold
// K {k0+8..15, k0+24..31} (2 packed per VGPR), column = m0 + (lane&15).
__device__ __forceinline__ bf16x16 load_b_frag(const __bf16* base, int k0, int half)
{
    const int o = k0 + 8 * half;
    bf16x8 lo = *(const bf16x8*)(base + o);
    bf16x8 hi = *(const bf16x8*)(base + o + 16);
    bf16x16 r;
#pragma unroll
    for (int i = 0; i < 8; ++i) { r[i] = lo[i]; r[i + 8] = hi[i]; }
    return r;
}

// ---------------------------------------------------------------------------
// Fused GEMM + masked cosine top-3 + weighted-average correspondence.
// One wave (32 threads) per 16 src rows; sweep all M in 16-col WMMA tiles.
// ---------------------------------------------------------------------------
__global__ __launch_bounds__(32)
void knn_kernel(const __bf16* __restrict__ fxT, const __bf16* __restrict__ fyT,
                const float* __restrict__ rscale,
                const unsigned char* __restrict__ mask1,
                float* __restrict__ pred)
{
    __shared__ float tile[16][17];
    __shared__ float mrg_r[32][3];
    __shared__ int   mrg_i[32][3];
    __shared__ float mrg_w[32][3];

    const int lane = threadIdx.x;
    const int half = lane >> 4;
    const int l16  = lane & 15;
    const int b    = blockIdx.y;
    const int n0   = blockIdx.x * 16;

    // A fragments for our 16 rows: loop-invariant, reused for all 784 tiles.
    const __bf16* Abase = fxT + ((size_t)(b * N_ + n0 + l16)) * C_;
    bf16x16 a0 = load_a_frag(Abase, 0,  half);
    bf16x16 a1 = load_a_frag(Abase, 32, half);

    const float NEG = -__builtin_inff();
    float r1 = NEG, r2 = NEG, r3 = NEG;
    int   i1 = 0,  i2 = 0,  i3 = 0;
    float w1 = 0.f, w2 = 0.f, w3 = 0.f;
    const int srow = l16;        // scan phase: lane handles row srow,
    const int scol = half * 8;   // columns scol..scol+7 of each tile
    const float* rs = rscale + (size_t)b * N_;

    for (int m0 = 0; m0 < N_; m0 += 16) {
        const __bf16* Bbase = fyT + ((size_t)(b * N_ + m0 + l16)) * C_;
        bf16x16 b0 = load_b_frag(Bbase, 0,  half);
        bf16x16 b1 = load_b_frag(Bbase, 32, half);
        if (m0 + 16 < N_)                      // pull next tile into WGP caches
            __builtin_prefetch(Bbase + 16 * C_, 0, 3);

        f32x8 acc = {0.f, 0.f, 0.f, 0.f, 0.f, 0.f, 0.f, 0.f};
        acc = __builtin_amdgcn_wmma_f32_16x16x32_bf16(false, a0, false, b0,
                                                      (short)0, acc, false, false);
        acc = __builtin_amdgcn_wmma_f32_16x16x32_bf16(false, a1, false, b1,
                                                      (short)0, acc, false, false);

        // C layout: acc[j] = sim[row j + 8*half][col l16]  ->  bounce via LDS.
        // Same-wave DS ops are in-order; no barrier needed.
#pragma unroll
        for (int j = 0; j < 8; ++j) tile[j + 8 * half][l16] = acc[j];

        // Batched reads: 8 LDS dwords (paired ds_load_2addr) + 2x b128 scales.
        float raw[8];
#pragma unroll
        for (int t = 0; t < 8; ++t) raw[t] = tile[srow][scol + t];
        const float4 s0 = *(const float4*)(rs + m0 + scol);
        const float4 s1 = *(const float4*)(rs + m0 + scol + 4);
        const float sc[8] = {s0.x, s0.y, s0.z, s0.w, s1.x, s1.y, s1.z, s1.w};

        float rank[8];
        float mx = NEG;
#pragma unroll
        for (int t = 0; t < 8; ++t) {
            rank[t] = (sc[t] > 0.0f) ? raw[t] * sc[t] : NEG;  // masked cosine
            mx = fmaxf(mx, rank[t]);
        }
        if (mx > r3) {                      // rare past warm-up: one execz skip
#pragma unroll
            for (int t = 0; t < 8; ++t) {
                const float rk = rank[t];
                if (rk > r3) {
                    const int m = m0 + scol + t;
                    if (rk > r1)      { r3=r2;i3=i2;w3=w2; r2=r1;i2=i1;w2=w1; r1=rk;i1=m;w1=raw[t]; }
                    else if (rk > r2) { r3=r2;i3=i2;w3=w2; r2=rk;i2=m;w2=raw[t]; }
                    else              {                    r3=rk;i3=m;w3=raw[t]; }
                }
            }
        }
    }

    // Merge the two column-halves per row (tie-break: lower index, as top_k).
    mrg_r[lane][0] = r1; mrg_r[lane][1] = r2; mrg_r[lane][2] = r3;
    mrg_i[lane][0] = i1; mrg_i[lane][1] = i2; mrg_i[lane][2] = i3;
    mrg_w[lane][0] = w1; mrg_w[lane][1] = w2; mrg_w[lane][2] = w3;

    if (lane < 16) {
        float R0 = NEG, R1 = NEG, R2 = NEG;
        int   I0 = 0,  I1 = 0,  I2 = 0;
        float W0 = 0.f, W1 = 0.f, W2 = 0.f;
#pragma unroll
        for (int q = 0; q < 6; ++q) {
            const int   src = lane + 16 * (q >= 3);
            const int   k   = q % 3;
            const float rr  = mrg_r[src][k];
            const int   ii  = mrg_i[src][k];
            const float ww  = mrg_w[src][k];
            const bool c0 = (rr > R0) || (rr == R0 && ii < I0);
            const bool c1 = (rr > R1) || (rr == R1 && ii < I1);
            const bool c2 = (rr > R2) || (rr == R2 && ii < I2);
            if (c0)      { R2=R1;I2=I1;W2=W1; R1=R0;I1=I0;W1=W0; R0=rr;I0=ii;W0=ww; }
            else if (c1) { R2=R1;I2=I1;W2=W1; R1=rr;I1=ii;W1=ww; }
            else if (c2) { R2=rr;I2=ii;W2=ww; }
        }
        const float den = W0 + W1 + W2;
        float ph = (float)(I0 / W_) * W0 + (float)(I1 / W_) * W1 + (float)(I2 / W_) * W2;
        float pw = (float)(I0 % W_) * W0 + (float)(I1 % W_) * W1 + (float)(I2 % W_) * W2;
        ph /= den; pw /= den;
        const int  n     = n0 + lane;
        const bool valid = mask1[(size_t)b * N_ + n] != 0;
        float* o = pred + ((size_t)b * N_ + n) * 2;
        o[0] = valid ? ph : -1.0f;
        o[1] = valid ? pw : -1.0f;
    }
}

// ---------------------------------------------------------------------------
// dst_pixel_group = pred_corr[round(src*H) clamped] / H
// ---------------------------------------------------------------------------
__global__ __launch_bounds__(256)
void gather_kernel(const float* __restrict__ spg, const float* __restrict__ pred,
                   float* __restrict__ out)
{
    int t = blockIdx.x * blockDim.x + threadIdx.x;
    if (t >= B_ * S_ * G_) return;
    const int b = t / (S_ * G_);
    const float y = spg[(size_t)t * 2 + 0] * (float)H_;
    const float x = spg[(size_t)t * 2 + 1] * (float)H_;
    int p0 = (int)rintf(y); p0 = p0 < 0 ? 0 : (p0 > H_ - 1 ? H_ - 1 : p0);
    int p1 = (int)rintf(x); p1 = p1 < 0 ? 0 : (p1 > W_ - 1 ? W_ - 1 : p1);
    const int lin = p0 * W_ + p1;
    const float* pc = pred + ((size_t)b * N_ + lin) * 2;
    const float inv = 1.0f / (float)H_;   // norm_h = slic.shape[1] = 112
    out[(size_t)t * 2 + 0] = pc[0] * inv;
    out[(size_t)t * 2 + 1] = pc[1] * inv;
}

extern "C" void kernel_launch(void* const* d_in, const int* in_sizes, int n_in,
                              void* d_out, int out_size, void* d_ws, size_t ws_size,
                              hipStream_t stream)
{
    (void)in_sizes; (void)n_in; (void)out_size; (void)ws_size;
    const float*         img1  = (const float*)d_in[0];
    const float*         img2  = (const float*)d_in[1];
    const unsigned char* mask1 = (const unsigned char*)d_in[2];
    const unsigned char* mask2 = (const unsigned char*)d_in[3];
    const float*         spg   = (const float*)d_in[4];
    // d_in[5] (slic) only contributes its shape (norm_h = 112), unused here.

    float* out = (float*)d_out;                 // [pred_corr | dst_pixel_group]

    char* ws = (char*)d_ws;
    const size_t bytes_fT = (size_t)B_ * N_ * C_ * sizeof(__bf16);  // 3,211,264
    __bf16* fxT    = (__bf16*)ws;
    __bf16* fyT    = (__bf16*)(ws + bytes_fT);
    float*  rscale = (float*)(ws + 2 * bytes_fT);

    {
        const int total = B_ * N_;
        prep_kernel<<<(total + 255) / 256, 256, 0, stream>>>(img1, img2, mask2,
                                                             fxT, fyT, rscale);
    }
    {
        dim3 grid(N_ / 16, B_);
        knn_kernel<<<grid, 32, 0, stream>>>(fxT, fyT, rscale, mask1, out);
    }
    {
        const int total = B_ * S_ * G_;
        gather_kernel<<<(total + 255) / 256, 256, 0, stream>>>(
            spg, out, out + (size_t)B_ * N_ * 2);
    }
}
